// AttEdgeAwareGNNV2_71339406786696
// MI455X (gfx1250) — matrix operand order repaired and checked by reading, not compiled, not run on hardware
//
#include <hip/hip_runtime.h>
#include <hip/hip_bf16.h>
#include <math.h>

// ---------------------------------------------------------------------------
// AttEdgeAwareGNNV2 for MI455X (gfx1250).
// Dense projections go through v_wmma_f32_16x16x32_bf16 (bf16 in, fp32 acc).
// Weights are prepacked ONCE per launch into per-lane WMMA B-fragment layout
// (bf16), so each GEMM wave loads a fragment with one coalesced 32B load
// instead of 16 strided scalar loads. Irregular segment-softmax / scatter
// parts use coalesced per-head threads with float atomics. Layer-2 edge
// update (y2) is dead code -> skipped.
// ---------------------------------------------------------------------------

#define NN 50000
#define NE 1600000
#define HD 32

typedef __attribute__((ext_vector_type(16))) __bf16 v16bf;
typedef __attribute__((ext_vector_type(8)))  float  v8f;

// Fragment = 32 lanes x 16 bf16 = 512 bf16 = 1KB. A matrix with KSTEPS
// k-steps packs into KSTEPS*2 fragments (2 column tiles of 16).
#define FRAG_ELEMS 512

static __device__ __forceinline__ v8f wmma_bf16(v16bf a, v16bf b, v8f c) {
  // 8 args: (neg_a, A, neg_b, B, c_mod, C, reuse_a, reuse_b)
  return __builtin_amdgcn_wmma_f32_16x16x32_bf16(false, a, false, b, (short)0, c,
                                                 false, false);
}

static __device__ __forceinline__ void atomicMaxFloat(float* addr, float v) {
  // Works for mixed-sign values with -inf init (ordered-int trick).
  if (v >= 0.0f) atomicMax((int*)addr, __float_as_int(v));
  else           atomicMin((unsigned int*)addr, __float_as_uint(v));
}

// ---------------- weight prepack: W[K x 32] f32 -> bf16 B-fragments ---------
// Fragment f = kt*2+nc; element (lane, j) = W[(kt*32 + (lane>>4)*8 + (j<8 ? j : j+8))*32
//                                             + nc*16 + (lane&15)].
// One wave packs everything (weights are tiny; runs once per launch).
__global__ void pack_weights(const float* __restrict__ W, __bf16* __restrict__ out,
                             int ksteps) {
  const int lane = threadIdx.x & 31;
  const int half = lane >> 4;
  const int n    = lane & 15;
  for (int kt = 0; kt < ksteps; ++kt)
    for (int nc = 0; nc < 2; ++nc) {
      __bf16* dstp = out + ((size_t)(kt * 2 + nc) * 32 + lane) * 16;
#pragma unroll
      for (int j = 0; j < 16; ++j)
        dstp[j] = (__bf16)W[(size_t)(kt * 32 + half * 8 + (j < 8 ? j : j + 8)) * 32 + nc * 16 + n];
    }
}

// ---------------- generic GEMM: C[M x 32] = A[M x 32*KSTEPS] @ W + bias -----
// One wave computes one 16-row tile. A row-major fp32, converted to bf16.
// Wpk = prepacked bf16 fragments (KSTEPS*2 fragments).
template <int KSTEPS>
__global__ void wmma_gemm_n32(const float* __restrict__ A,
                              const __bf16* __restrict__ Wpk,
                              const float* __restrict__ bias,
                              float* __restrict__ C, int ntiles) {
  const int wave = threadIdx.x >> 5;
  const int lane = threadIdx.x & 31;
  const int tile = blockIdx.x * (blockDim.x >> 5) + wave;
  if (tile >= ntiles) return;               // wave-uniform: EXEC stays all-1s
  const int half = lane >> 4;               // 0: lanes 0-15, 1: lanes 16-31
  const int n    = lane & 15;

  // B fragments: one coalesced 32B vector load per fragment per lane.
  v16bf bfrag[KSTEPS][2];
#pragma unroll
  for (int kt = 0; kt < KSTEPS; ++kt)
#pragma unroll
    for (int nc = 0; nc < 2; ++nc)
      bfrag[kt][nc] =
          *(const v16bf*)(Wpk + ((size_t)(kt * 2 + nc) * 32 + lane) * 16);

  v8f acc0, acc1;
  const float b0 = bias[n], b1 = bias[16 + n];
#pragma unroll
  for (int r = 0; r < 8; ++r) { acc0[r] = b0; acc1[r] = b1; }

  // A fragment: lane half=0 -> K [0..7]+[16..23]; half=1 -> [8..15]+[24..31]
  const float* arow = A + (size_t)(tile * 16 + n) * (KSTEPS * 32);
#pragma unroll
  for (int kt = 0; kt < KSTEPS; ++kt) {
    v16bf af;
    const int kb = kt * 32 + half * 8;
#pragma unroll
    for (int j = 0; j < 8; ++j) {
      af[j]     = (__bf16)arow[kb + j];
      af[8 + j] = (__bf16)arow[kb + 16 + j];
    }
    acc0 = wmma_bf16(af, bfrag[kt][0], acc0);
    acc1 = wmma_bf16(af, bfrag[kt][1], acc1);
  }

  const size_t mbase = (size_t)tile * 16 + half * 8;
#pragma unroll
  for (int r = 0; r < 8; ++r) {
    C[(mbase + r) * 32 + n]      = acc0[r];
    C[(mbase + r) * 32 + 16 + n] = acc1[r];
  }
}

// -------- edge update GEMM: Y[E x 32] = [hs || hd || he] @ Weo + beo --------
// A rows (K=96) are gathered on the fly from hn[src], hn[dst], he[e].
__global__ void wmma_edgeout(const float* __restrict__ hn,
                             const float* __restrict__ he,
                             const int* __restrict__ src,
                             const int* __restrict__ dst,
                             const __bf16* __restrict__ Wpk,
                             const float* __restrict__ beo,
                             float* __restrict__ Y, int ntiles) {
  const int wave = threadIdx.x >> 5;
  const int lane = threadIdx.x & 31;
  const int tile = blockIdx.x * (blockDim.x >> 5) + wave;
  if (tile >= ntiles) return;
  const int half = lane >> 4;
  const int n    = lane & 15;

  v16bf bfrag[3][2];
#pragma unroll
  for (int kt = 0; kt < 3; ++kt)
#pragma unroll
    for (int nc = 0; nc < 2; ++nc)
      bfrag[kt][nc] =
          *(const v16bf*)(Wpk + ((size_t)(kt * 2 + nc) * 32 + lane) * 16);

  v8f acc0, acc1;
  const float b0 = beo[n], b1 = beo[16 + n];
#pragma unroll
  for (int r = 0; r < 8; ++r) { acc0[r] = b0; acc1[r] = b1; }

  const int e = tile * 16 + n;
  const float* rows[3] = { hn + (size_t)src[e] * 32,
                           hn + (size_t)dst[e] * 32,
                           he + (size_t)e      * 32 };
#pragma unroll
  for (int kt = 0; kt < 3; ++kt) {
    const float* arow = rows[kt];
    v16bf af;
    const int kb = half * 8;
#pragma unroll
    for (int j = 0; j < 8; ++j) {
      af[j]     = (__bf16)arow[kb + j];
      af[8 + j] = (__bf16)arow[kb + 16 + j];
    }
    acc0 = wmma_bf16(af, bfrag[kt][0], acc0);
    acc1 = wmma_bf16(af, bfrag[kt][1], acc1);
  }

  const size_t mbase = (size_t)tile * 16 + half * 8;
#pragma unroll
  for (int r = 0; r < 8; ++r) {
    Y[(mbase + r) * 32 + n]      = acc0[r];
    Y[(mbase + r) * 32 + 16 + n] = acc1[r];
  }
}

// ---------------- irregular passes (one thread per edge*head) ---------------
__global__ void attn_logits(const float* __restrict__ hn, const float* __restrict__ he,
                            const int* __restrict__ src, const int* __restrict__ dst,
                            const float* __restrict__ a_s, const float* __restrict__ a_d,
                            const float* __restrict__ a_e,
                            float* __restrict__ logit, float* __restrict__ mmax) {
  const unsigned t = blockIdx.x * blockDim.x + threadIdx.x;
  if (t >= (unsigned)NE * 4u) return;
  const int e = t >> 2, h = t & 3;
  const int s = src[e], d = dst[e];
  const float* ps = hn + (size_t)s * 32 + h * 8;
  const float* pd = hn + (size_t)d * 32 + h * 8;
  const float* pe = he + (size_t)e * 32 + h * 8;
  float acc = 0.f;
#pragma unroll
  for (int j = 0; j < 8; ++j)
    acc += ps[j] * a_s[h * 8 + j] + pd[j] * a_d[h * 8 + j] + pe[j] * a_e[h * 8 + j];
  acc = acc > 0.f ? acc : 0.2f * acc;       // leaky_relu(0.2)
  logit[(size_t)e * 4 + h] = acc;
  atomicMaxFloat(&mmax[(size_t)d * 4 + h], acc);
}

__global__ void attn_denom(const int* __restrict__ dst, const float* __restrict__ logit,
                           const float* __restrict__ mmax, float* __restrict__ denom) {
  const unsigned t = blockIdx.x * blockDim.x + threadIdx.x;
  if (t >= (unsigned)NE * 4u) return;
  const int e = t >> 2, h = t & 3;
  const int d = dst[e];
  atomicAdd(&denom[(size_t)d * 4 + h],
            __expf(logit[(size_t)e * 4 + h] - mmax[(size_t)d * 4 + h]));
}

__global__ void attn_scatter(const float* __restrict__ hn, const float* __restrict__ he,
                             const int* __restrict__ src, const int* __restrict__ dst,
                             const float* __restrict__ logit, const float* __restrict__ mmax,
                             const float* __restrict__ denom, float* __restrict__ X) {
  const unsigned t = blockIdx.x * blockDim.x + threadIdx.x;
  if (t >= (unsigned)NE * 4u) return;
  const int e = t >> 2, h = t & 3;
  const int s = src[e], d = dst[e];
  const float alpha = __expf(logit[(size_t)e * 4 + h] - mmax[(size_t)d * 4 + h]) /
                      (denom[(size_t)d * 4 + h] + 1e-9f);
  const float* ps = hn + (size_t)s * 32 + h * 8;
  const float* pe = he + (size_t)e * 32 + h * 8;
#pragma unroll
  for (int j = 0; j < 8; ++j)
    atomicAdd(&X[(size_t)d * 32 + h * 8 + j], alpha * (ps[j] + pe[j]));
}

// ---------------- elementwise helpers ---------------------------------------
__global__ void fill_kernel(float* __restrict__ p, float v, size_t n) {
  const size_t i = (size_t)blockIdx.x * blockDim.x + threadIdx.x;
  if (i < n) p[i] = v;
}

__global__ void row_softmax32(float* __restrict__ X, int nrows) {
  const int r = blockIdx.x * blockDim.x + threadIdx.x;
  if (r >= nrows) return;
  float* p = X + (size_t)r * 32;
  float v[32];
  float m = -INFINITY;
#pragma unroll
  for (int j = 0; j < 32; ++j) { v[j] = p[j]; m = fmaxf(m, v[j]); }
  float s = 0.f;
#pragma unroll
  for (int j = 0; j < 32; ++j) { v[j] = __expf(v[j] - m); s += v[j]; }
  const float inv = 1.f / s;
#pragma unroll
  for (int j = 0; j < 32; ++j) p[j] = v[j] * inv;
}

__global__ void relu_inplace(float* __restrict__ p, size_t n) {
  const size_t i = (size_t)blockIdx.x * blockDim.x + threadIdx.x;
  if (i < n) p[i] = fmaxf(p[i], 0.f);
}

__global__ void head_kernel(const float* __restrict__ X, const float* __restrict__ Wl,
                            const float* __restrict__ bl, float* __restrict__ out) {
  const int i = blockIdx.x * blockDim.x + threadIdx.x;
  if (i >= NN) return;
  const float* row = X + (size_t)i * 32;
  float acc = bl[0];
#pragma unroll
  for (int j = 0; j < 32; ++j) acc += row[j] * Wl[j];
  out[i] = acc;
}

// ---------------------------------------------------------------------------
extern "C" void kernel_launch(void* const* d_in, const int* in_sizes, int n_in,
                              void* d_out, int out_size, void* d_ws, size_t ws_size,
                              hipStream_t stream) {
  const float* nodes = (const float*)d_in[0];   // [50000,128]
  const float* edges = (const float*)d_in[1];   // [1.6M,32]
  const int*   ei    = (const int*)d_in[2];     // [2,1.6M]
  const int* src = ei;
  const int* dst = ei + NE;
  const float *Wn1 = (const float*)d_in[3],  *bn1 = (const float*)d_in[4];
  const float *We1 = (const float*)d_in[5],  *be1 = (const float*)d_in[6];
  const float *as1 = (const float*)d_in[7],  *ad1 = (const float*)d_in[8],
              *ae1 = (const float*)d_in[9];
  const float *Weo1 = (const float*)d_in[10], *beo1 = (const float*)d_in[11];
  const float *Wn2 = (const float*)d_in[12], *bn2 = (const float*)d_in[13];
  const float *We2 = (const float*)d_in[14], *be2 = (const float*)d_in[15];
  const float *as2 = (const float*)d_in[16], *ad2 = (const float*)d_in[17],
              *ae2 = (const float*)d_in[18];
  // d_in[19]=Weo2, d_in[20]=beo2 unused: y2 never reaches the output.
  const float *Wl = (const float*)d_in[21], *bl = (const float*)d_in[22];
  float* out = (float*)d_out;

  // Workspace layout (fp32 region then bf16 weight arena), ~456 MB total.
  float* ws    = (float*)d_ws;
  float* hn    = ws;                         // [N,32]   projected nodes
  float* he    = hn    + (size_t)NN * 32;    // [E,32]   projected edges
  float* logit = he    + (size_t)NE * 32;    // [E,4]
  float* mmax  = logit + (size_t)NE * 4;     // [N,4]
  float* denom = mmax  + (size_t)NN * 4;     // [N,4]
  float* x1    = denom + (size_t)NN * 4;     // [N,32]   layer-1 node out
  float* y1    = x1    + (size_t)NN * 32;    // [E,32]   layer-1 edge out
  float* x2    = y1    + (size_t)NE * 32;    // [N,32]   layer-2 node out
  // bf16 weight-fragment arena (all offsets are 32B aligned).
  __bf16* wpkWn1 = (__bf16*)(x2 + (size_t)NN * 32);   // 4 ksteps -> 8 frags
  __bf16* wpkWe1 = wpkWn1 + 8 * FRAG_ELEMS;           // 2 frags
  __bf16* wpkWeo = wpkWe1 + 2 * FRAG_ELEMS;           // 6 frags
  __bf16* wpkWn2 = wpkWeo + 6 * FRAG_ELEMS;           // 2 frags
  __bf16* wpkWe2 = wpkWn2 + 2 * FRAG_ELEMS;           // 2 frags

  const int ntilesN = NN / 16;               // 3125 (exact)
  const int ntilesE = NE / 16;               // 100000 (exact)
  const dim3 blk(256);
  const dim3 one(32);
  const dim3 gN_gemm((ntilesN + 7) / 8);     // 8 waves / block
  const dim3 gE_gemm(ntilesE / 8);
  const dim3 gEH(((unsigned)NE * 4u + 255u) / 256u);
  const dim3 gNode((NN + 255) / 256);
  const dim3 gEdgeRows((NE + 255) / 256);
  auto fill = [&](float* p, float v, size_t n) {
    fill_kernel<<<dim3((unsigned)((n + 255) / 256)), blk, 0, stream>>>(p, v, n);
  };

  // ---------------- prepack all GEMM weights (runs once, ~20KB) -------------
  pack_weights<<<dim3(1), one, 0, stream>>>(Wn1,  wpkWn1, 4);
  pack_weights<<<dim3(1), one, 0, stream>>>(We1,  wpkWe1, 1);
  pack_weights<<<dim3(1), one, 0, stream>>>(Weo1, wpkWeo, 3);
  pack_weights<<<dim3(1), one, 0, stream>>>(Wn2,  wpkWn2, 1);
  pack_weights<<<dim3(1), one, 0, stream>>>(We2,  wpkWe2, 1);

  // ---------------- Layer 1 (nl = softmax) ----------------
  wmma_gemm_n32<4><<<gN_gemm, blk, 0, stream>>>(nodes, wpkWn1, bn1, hn, ntilesN);
  wmma_gemm_n32<1><<<gE_gemm, blk, 0, stream>>>(edges, wpkWe1, be1, he, ntilesE);
  fill(mmax, -INFINITY, (size_t)NN * 4);
  fill(denom, 0.f, (size_t)NN * 4);
  fill(x1, 0.f, (size_t)NN * 32);
  attn_logits<<<gEH, blk, 0, stream>>>(hn, he, src, dst, as1, ad1, ae1, logit, mmax);
  attn_denom <<<gEH, blk, 0, stream>>>(dst, logit, mmax, denom);
  attn_scatter<<<gEH, blk, 0, stream>>>(hn, he, src, dst, logit, mmax, denom, x1);
  row_softmax32<<<gNode, blk, 0, stream>>>(x1, NN);
  wmma_edgeout<<<gE_gemm, blk, 0, stream>>>(hn, he, src, dst, wpkWeo, beo1, y1, ntilesE);
  row_softmax32<<<gEdgeRows, blk, 0, stream>>>(y1, NE);

  // ---------------- Layer 2 (nl = relu; y2 dead -> skipped) ----------------
  wmma_gemm_n32<1><<<gN_gemm, blk, 0, stream>>>(x1, wpkWn2, bn2, hn, ntilesN);
  wmma_gemm_n32<1><<<gE_gemm, blk, 0, stream>>>(y1, wpkWe2, be2, he, ntilesE);
  fill(mmax, -INFINITY, (size_t)NN * 4);
  fill(denom, 0.f, (size_t)NN * 4);
  fill(x2, 0.f, (size_t)NN * 32);
  attn_logits<<<gEH, blk, 0, stream>>>(hn, he, src, dst, as2, ad2, ae2, logit, mmax);
  attn_denom <<<gEH, blk, 0, stream>>>(dst, logit, mmax, denom);
  attn_scatter<<<gEH, blk, 0, stream>>>(hn, he, src, dst, logit, mmax, denom, x2);
  relu_inplace<<<dim3((unsigned)(((size_t)NN * 32 + 255) / 256)), blk, 0, stream>>>(
      x2, (size_t)NN * 32);

  // ---------------- Head: out = x2 @ Wl + bl ----------------
  head_kernel<<<gNode, blk, 0, stream>>>(x2, Wl, bl, out);
  (void)in_sizes; (void)n_in; (void)out_size; (void)ws_size;
}
